// GCN_36593121362339
// MI455X (gfx1250) — matrix operand order repaired and checked by reading, not compiled, not run on hardware
//
#include <hip/hip_runtime.h>

// ---------------------------------------------------------------------------
// GCN (3x GCNConv+ReLU, 1x GCNConv) for gfx1250 (MI455X).
// GEMMs run on v_wmma_f32_16x16x32_bf16 with split-bf16 (hi+lo) operands for
// ~fp32 accuracy (3 WMMAs per 32-wide k-tile). GEMM epilogue fuses the
// self-loop aggregation init (agg = hw * dis^2), saving a full 51 MB f32
// pass per layer. Edge aggregation is fp32 gather + global_atomic_add_f32
// scatter; the 51 MB hw/agg working set is resident in the 192 MB L2, so
// atomics resolve at L2 speed.
// ---------------------------------------------------------------------------

typedef __attribute__((ext_vector_type(16))) __bf16 v16bf;
typedef __attribute__((ext_vector_type(8)))  float  v8f;
typedef __attribute__((ext_vector_type(4)))  float  v4f;

static constexpr int NN  = 50000;   // nodes (50000 = 3125 * 16, no M tail)
static constexpr int EE  = 800000;  // edges
static constexpr int FIN = 256;     // input features
static constexpr int HH  = 128;     // hidden
static constexpr int NC  = 16;      // classes

// ---------------- small prep kernels ----------------------------------------

__global__ void k_fill1(float* p, int n) {
    int i = blockIdx.x * blockDim.x + threadIdx.x;
    if (i < n) p[i] = 1.0f;                       // self-loop contributes 1 to deg
}

__global__ void k_degcount(const int* __restrict__ dst, float* deg, int e) {
    int i = blockIdx.x * blockDim.x + threadIdx.x;
    if (i < e) atomicAdd(&deg[dst[i]], 1.0f);
}

__global__ void k_rsqrt_inplace(float* p, int n) {  // deg -> deg^-1/2 (deg>=1)
    int i = blockIdx.x * blockDim.x + threadIdx.x;
    if (i < n) p[i] = rsqrtf(p[i]);
}

__global__ void k_norm(const int* __restrict__ src, const int* __restrict__ dst,
                       const float* __restrict__ dis, float* normE, int e) {
    int i = blockIdx.x * blockDim.x + threadIdx.x;
    if (i < e) normE[i] = dis[src[i]] * dis[dst[i]];
}

// split fp32 -> bf16 hi + bf16 lo (x ~= hi + lo)
__global__ void k_split(const float* __restrict__ x, __bf16* hi, __bf16* lo, int n) {
    int i = blockIdx.x * blockDim.x + threadIdx.x;
    if (i >= n) return;
    float v = x[i];
    __bf16 h = (__bf16)v;
    hi[i] = h;
    lo[i] = (__bf16)(v - (float)h);
}

// Pack weight W[K x OUT] (row-major) into the per-lane WMMA B-fragment layout:
// element i of lane L (tile kt,nt): K = kt*32 + (L/16)*16 + i, Ncol = nt*16 + L%16.
__global__ void k_packw(const float* __restrict__ W, __bf16* hi, __bf16* lo,
                        int K, int OUT) {
    int idx = blockIdx.x * blockDim.x + threadIdx.x;
    if (idx >= K * OUT) return;
    int i      = idx & 15;
    int lane   = (idx >> 4) & 31;
    int t      = idx >> 9;                 // kt*ntiles + nt
    int ntiles = OUT >> 4;
    int kt = t / ntiles, nt = t - kt * ntiles;
    int n  = nt * 16 + (lane & 15);
    int hf = lane >> 4;
    int k  = kt * 32 + hf * 16 + i;
    float v  = W[(size_t)k * OUT + n];
    __bf16 h = (__bf16)v;
    hi[idx] = h;
    lo[idx] = (__bf16)(v - (float)h);
}

// ---------------- WMMA GEMM: hw[M x OUT] = A[M x K] * B[K x OUT] ------------
// A given as split bf16 hi/lo, row-major. B given pre-packed (hi/lo).
// One wave -> one 16x16 tile. 3 WMMAs per 32-wide k-tile (hi*hi+hi*lo+lo*hi).
// Fused epilogue: agg[r][c] = acc * dis[r]^2 (self-loop init for aggregation).

__global__ __launch_bounds__(256)
void k_gemm_bf16x2(const __bf16* __restrict__ Ahi, const __bf16* __restrict__ Alo,
                   const __bf16* __restrict__ Bhi, const __bf16* __restrict__ Blo,
                   float* __restrict__ Cout, float* __restrict__ AggInit,
                   const float* __restrict__ dis, int M, int K, int OUT) {
    const int ntiles = OUT >> 4;
    const int mtiles = M >> 4;
    const int ktiles = K >> 5;
    int wave = blockIdx.x * (blockDim.x >> 5) + (threadIdx.x >> 5);
    if (wave >= mtiles * ntiles) return;          // wave-uniform: EXEC stays full
    int mt = wave / ntiles, nt = wave - mt * ntiles;

    int lane = threadIdx.x & 31;
    int mloc = lane & 15;
    int hf   = lane >> 4;

    const size_t arow = (size_t)(mt * 16 + mloc) * K;
    const __bf16* aph = Ahi + arow;
    const __bf16* apl = Alo + arow;
    const size_t boff    = ((size_t)nt * 32 + lane) * 16;
    const size_t bstride = (size_t)ntiles * 512;  // elements per k-tile of packed B

    union Frag { v4f f[2]; v16bf bf; };
    v8f acc = {};

    for (int kt = 0; kt < ktiles; ++kt) {
        int k0 = kt * 32 + hf * 16;
        // pull next A k-tile toward the WGP (speculative; dropped if OOB)
        __builtin_prefetch(aph + k0 + 32, 0, 3);

        Frag ah, al, bh, bl;
        ah.f[0] = *(const v4f*)(aph + k0);
        ah.f[1] = *(const v4f*)(aph + k0 + 8);
        al.f[0] = *(const v4f*)(apl + k0);
        al.f[1] = *(const v4f*)(apl + k0 + 8);
        const __bf16* bhp = Bhi + boff + (size_t)kt * bstride;
        const __bf16* blp = Blo + boff + (size_t)kt * bstride;
        bh.f[0] = *(const v4f*)(bhp);
        bh.f[1] = *(const v4f*)(bhp + 8);
        bl.f[0] = *(const v4f*)(blp);
        bl.f[1] = *(const v4f*)(blp + 8);

        acc = __builtin_amdgcn_wmma_f32_16x16x32_bf16(false, ah.bf, false, bh.bf,
                                                      (short)0, acc, false, false);
        acc = __builtin_amdgcn_wmma_f32_16x16x32_bf16(false, ah.bf, false, bl.bf,
                                                      (short)0, acc, false, false);
        acc = __builtin_amdgcn_wmma_f32_16x16x32_bf16(false, al.bf, false, bh.bf,
                                                      (short)0, acc, false, false);
    }

    // D layout: VGPR j -> row hf*8 + j, col = lane%16
    const int rbase = mt * 16 + hf * 8;
    const size_t coff = (size_t)nt * 16 + mloc;
#pragma unroll
    for (int j = 0; j < 8; ++j) {
        size_t o = (size_t)(rbase + j) * OUT + coff;
        float v = acc[j];
        Cout[o] = v;
        float s = dis[rbase + j];
        AggInit[o] = v * (s * s);        // self-loop term seeds the aggregation
    }
}

// ---------------- aggregation ----------------------------------------------

// one wave per edge, 4 floats/lane (F = 128)
__global__ void k_edge128(const int* __restrict__ src, const int* __restrict__ dst,
                          const float* __restrict__ normE, const float* __restrict__ hw,
                          float* __restrict__ agg, int e) {
    int gid = blockIdx.x * blockDim.x + threadIdx.x;
    int eid = gid >> 5;
    if (eid >= e) return;
    int lane = gid & 31;
    int s = src[eid], d = dst[eid];
    float nm = normE[eid];
    float4 v = ((const float4*)(hw + (size_t)s * 128))[lane];
    float* o = agg + (size_t)d * 128 + lane * 4;
    atomicAdd(o + 0, v.x * nm);
    atomicAdd(o + 1, v.y * nm);
    atomicAdd(o + 2, v.z * nm);
    atomicAdd(o + 3, v.w * nm);
}

// one thread per (edge, feature), F = 16
__global__ void k_edge16(const int* __restrict__ src, const int* __restrict__ dst,
                         const float* __restrict__ normE, const float* __restrict__ hw,
                         float* __restrict__ agg, int e) {
    int idx = blockIdx.x * blockDim.x + threadIdx.x;
    if (idx >= e * 16) return;
    int eid = idx >> 4, f = idx & 15;
    int s = src[eid], d = dst[eid];
    atomicAdd(agg + (size_t)d * 16 + f, hw[(size_t)s * 16 + f] * normE[eid]);
}

// h = relu(agg + b); write split bf16 for next GEMM; optional fp32 copy (latent)
__global__ void k_relu_split(const float* __restrict__ agg, const float* __restrict__ bias,
                             __bf16* hi, __bf16* lo, float* f32out, int n, int F) {
    int idx = blockIdx.x * blockDim.x + threadIdx.x;
    if (idx >= n * F) return;
    int f = idx % F;
    float v = agg[idx] + bias[f];
    v = v > 0.0f ? v : 0.0f;
    __bf16 h = (__bf16)v;
    hi[idx] = h;
    lo[idx] = (__bf16)(v - (float)h);
    if (f32out) f32out[idx] = v;
}

__global__ void k_biasout(const float* __restrict__ agg, const float* __restrict__ bias,
                          float* __restrict__ out, int n, int F) {
    int idx = blockIdx.x * blockDim.x + threadIdx.x;
    if (idx >= n * F) return;
    out[idx] = agg[idx] + bias[idx % F];
}

// ---------------- host launch ----------------------------------------------

extern "C" void kernel_launch(void* const* d_in, const int* in_sizes, int n_in,
                              void* d_out, int out_size, void* d_ws, size_t ws_size,
                              hipStream_t stream) {
    const float* x   = (const float*)d_in[0];
    const int*   ei  = (const int*)d_in[1];
    const float* W1  = (const float*)d_in[2];
    const float* b1  = (const float*)d_in[3];
    const float* Wh0 = (const float*)d_in[4];
    const float* bh0 = (const float*)d_in[5];
    const float* Wh1 = (const float*)d_in[6];
    const float* bh1 = (const float*)d_in[7];
    const float* W2  = (const float*)d_in[8];
    const float* b2  = (const float*)d_in[9];

    const int* src = ei;        // edge_index[0]
    const int* dst = ei + EE;   // edge_index[1]

    float* out  = (float*)d_out;              // [NN x NC]
    float* xlat = out + (size_t)NN * NC;      // [NN x HH] latent (h3)

    char* p = (char*)d_ws;
    auto take = [&](size_t bytes) -> char* {
        char* r = p;
        p += (bytes + 255) & ~(size_t)255;
        return r;
    };
    float*  dis   = (float*)take((size_t)NN * 4);          // deg -> deg^-1/2
    float*  normE = (float*)take((size_t)EE * 4);
    __bf16* x_hi  = (__bf16*)take((size_t)NN * FIN * 2);
    __bf16* x_lo  = (__bf16*)take((size_t)NN * FIN * 2);
    __bf16* h_hi  = (__bf16*)take((size_t)NN * HH * 2);
    __bf16* h_lo  = (__bf16*)take((size_t)NN * HH * 2);
    float*  hw    = (float*)take((size_t)NN * HH * 4);
    float*  agg   = (float*)take((size_t)NN * HH * 4);
    __bf16* W1h   = (__bf16*)take((size_t)FIN * HH * 2);
    __bf16* W1l   = (__bf16*)take((size_t)FIN * HH * 2);
    __bf16* Wh0h  = (__bf16*)take((size_t)HH * HH * 2);
    __bf16* Wh0l  = (__bf16*)take((size_t)HH * HH * 2);
    __bf16* Wh1h  = (__bf16*)take((size_t)HH * HH * 2);
    __bf16* Wh1l  = (__bf16*)take((size_t)HH * HH * 2);
    __bf16* W2h   = (__bf16*)take((size_t)HH * NC * 2);
    __bf16* W2l   = (__bf16*)take((size_t)HH * NC * 2);

    auto nb = [](long long n) { return dim3((unsigned)((n + 255) / 256)); };
    const dim3 blk(256);

    // normalization (shared by all layers)
    k_fill1<<<nb(NN), blk, 0, stream>>>(dis, NN);
    k_degcount<<<nb(EE), blk, 0, stream>>>(dst, dis, EE);
    k_rsqrt_inplace<<<nb(NN), blk, 0, stream>>>(dis, NN);
    k_norm<<<nb(EE), blk, 0, stream>>>(src, dst, dis, normE, EE);

    // operand prep
    k_split<<<nb((long long)NN * FIN), blk, 0, stream>>>(x, x_hi, x_lo, NN * FIN);
    k_packw<<<nb(FIN * HH), blk, 0, stream>>>(W1, W1h, W1l, FIN, HH);
    k_packw<<<nb(HH * HH), blk, 0, stream>>>(Wh0, Wh0h, Wh0l, HH, HH);
    k_packw<<<nb(HH * HH), blk, 0, stream>>>(Wh1, Wh1h, Wh1l, HH, HH);
    k_packw<<<nb(HH * NC), blk, 0, stream>>>(W2, W2h, W2l, HH, NC);

    const int mtiles = NN / 16;  // 3125 exactly
    auto gblk = [&](int ntiles) { return dim3((unsigned)((mtiles * ntiles + 7) / 8)); };

    // layer 1: x[N,256] @ W1 -> hw[N,128] (+ agg seeded); aggregate; relu
    k_gemm_bf16x2<<<gblk(HH / 16), blk, 0, stream>>>(x_hi, x_lo, W1h, W1l, hw, agg, dis, NN, FIN, HH);
    k_edge128<<<nb((long long)EE * 32), blk, 0, stream>>>(src, dst, normE, hw, agg, EE);
    k_relu_split<<<nb((long long)NN * HH), blk, 0, stream>>>(agg, b1, h_hi, h_lo, nullptr, NN, HH);

    // layer 2
    k_gemm_bf16x2<<<gblk(HH / 16), blk, 0, stream>>>(h_hi, h_lo, Wh0h, Wh0l, hw, agg, dis, NN, HH, HH);
    k_edge128<<<nb((long long)EE * 32), blk, 0, stream>>>(src, dst, normE, hw, agg, EE);
    k_relu_split<<<nb((long long)NN * HH), blk, 0, stream>>>(agg, bh0, h_hi, h_lo, nullptr, NN, HH);

    // layer 3 (latent output in fp32)
    k_gemm_bf16x2<<<gblk(HH / 16), blk, 0, stream>>>(h_hi, h_lo, Wh1h, Wh1l, hw, agg, dis, NN, HH, HH);
    k_edge128<<<nb((long long)EE * 32), blk, 0, stream>>>(src, dst, normE, hw, agg, EE);
    k_relu_split<<<nb((long long)NN * HH), blk, 0, stream>>>(agg, bh1, h_hi, h_lo, xlat, NN, HH);

    // output layer: h3[N,128] @ W2 -> [N,16] (+ agg seeded); aggregate; + b2
    k_gemm_bf16x2<<<gblk(NC / 16), blk, 0, stream>>>(h_hi, h_lo, W2h, W2l, hw, agg, dis, NN, HH, NC);
    k_edge16<<<nb((long long)EE * 16), blk, 0, stream>>>(src, dst, normE, hw, agg, EE);
    k_biasout<<<nb((long long)NN * NC), blk, 0, stream>>>(agg, b2, out, NN, NC);
}
